// Recurrence_27659589386662
// MI455X (gfx1250) — matrix operand order repaired and checked by reading, not compiled, not run on hardware
//
#include <hip/hip_runtime.h>
#include <hip/hip_bf16.h>

typedef __attribute__((ext_vector_type(16))) _Float16 v16h;
typedef __attribute__((ext_vector_type(8)))  float    v8f;

#define Tn 20
#define Nn 96
#define Ln 20
#define Hn 256
#define Cn 64
#define NAn 40
#define NOn 5
#define G3H 768          // 3*H
#define Bte 1920         // L*N batch of te-GRU
#define GR  76800        // L*L*N*2 rows of G

// ---------------- WMMA fragment helpers (CDNA5 16-bit layouts) ----------------
// A (16x32 MxK) from LDS row-major tile:
// lanes 0-15 -> M=lane, lanes 16-31 -> M=lane-16; VGPR v holds K = (v<4?0:16)+khalf+2(v&3)+{0,1}
__device__ inline void load_a_frag(const _Float16* As, int ldk, int kb, int lane, v16h& a) {
  int m = lane & 15;
  int khalf = (lane < 16) ? 0 : 8;
  const _Float16* row = As + (size_t)m * ldk;
#pragma unroll
  for (int v = 0; v < 8; ++v) {
    int kk = kb + ((v < 4) ? 0 : 16) + khalf + 2 * (v & 3);
    a[2 * v]     = row[kk];
    a[2 * v + 1] = row[kk + 1];
  }
}
// B fragment from PACKED buffer: layout [ct][kt][lane][16 halves] -> one 32B vector load per lane.
__device__ inline v16h load_b_packed(const _Float16* Bf, int KT, int kt, int ct, int lane) {
  return *reinterpret_cast<const v16h*>(Bf + (((size_t)ct * KT + kt) * 32 + lane) * 16);
}

__device__ inline float sigm(float x) { return 1.f / (1.f + __expf(-x)); }

// CDNA5 async memory->LDS copy (ASYNCcnt-tracked), 16 bytes per issuing lane.
__device__ inline void async_b128_to_lds(unsigned lds_off, const void* gaddr) {
  asm volatile("global_load_async_to_lds_b128 %0, %1, off"
               :: "v"(lds_off), "v"(gaddr) : "memory");
}
__device__ inline void wait_async0() {
  asm volatile("s_wait_asynccnt 0x0" ::: "memory");
}

// ---------------- B-weight packer: W (fp32) -> fragment-layout f16 ----------------
// B logical is (K x N). If TRANS: B[k][n] = W[n*K + k] (W is N x K, e.g. Wih -> Wih^T),
// else B[k][n] = W[k*N + n].
template <bool TRANS>
__global__ void pack_b(const float* W, _Float16* dst, int K, int N) {
  int idx = blockIdx.x * 256 + threadIdx.x;
  if (idx >= K * N) return;
  int e    = idx & 15;
  int lane = (idx >> 4) & 31;
  int f    = idx >> 9;
  int KT   = K >> 5;
  int kt   = f % KT;
  int ct   = f / KT;
  int n    = lane & 15;
  int koff = (lane < 16) ? 0 : 8;
  int kk   = kt * 32 + koff + ((e < 8) ? e : 8 + e); // e<8: K=koff+e; e>=8: K=16+koff+(e-8)
  int col  = ct * 16 + n;
  float v  = TRANS ? W[(size_t)col * K + kk] : W[(size_t)kk * N + col];
  dst[idx] = (_Float16)v;
}

// ---------------- generic WMMA GEMM: C = act(A(MxK) * Bpacked + bias) ----------------
// grid.x = M/16, block = 256 (8 waves). K multiple of 32 (<=320), N multiple of 16.
// f16-A path stages the contiguous 16xK tile via async global->LDS (no VGPR round-trip).
template <bool A16F, bool C16F, bool RELU>
__global__ void wmma_gemm(const float* A32, const _Float16* A16, const _Float16* Bf,
                          const float* bias, float* C32, _Float16* C16,
                          int K, int N) {
  __shared__ _Float16 As[16 * 320];
  const int tid = threadIdx.x;
  const int rowBase = blockIdx.x * 16;
  if (A16F) {
    unsigned ldsb = (unsigned)(uintptr_t)(&As[0]);
    const char* g = (const char*)(A16 + (size_t)rowBase * K);
    const int bytes = 16 * K * 2;
    for (int off = tid * 16; off < bytes; off += 256 * 16)
      async_b128_to_lds(ldsb + off, g + off);
    wait_async0();
  } else {
    const int tot = 16 * K;
    for (int idx = tid; idx < tot; idx += 256) {
      int r = idx / K, c = idx - r * K;
      As[idx] = (_Float16)A32[(size_t)(rowBase + r) * K + c];
    }
  }
  __syncthreads();
  const int lane = tid & 31, wave = tid >> 5;
  const int ntiles = N >> 4;
  const int KT = K >> 5;
  for (int ct = wave; ct < ntiles; ct += 8) {
    v8f acc = {0.f, 0.f, 0.f, 0.f, 0.f, 0.f, 0.f, 0.f};
    for (int kt = 0; kt < KT; ++kt) {
      v16h a;
      load_a_frag(As, K, kt * 32, lane, a);
      v16h b = load_b_packed(Bf, KT, kt, ct, lane);
      acc = __builtin_amdgcn_wmma_f32_16x16x32_f16(false, a, false, b, (short)0, acc, false, false);
    }
    const int n = lane & 15;
    const int moff = (lane < 16) ? 0 : 8;
    const float bv = bias[ct * 16 + n];
#pragma unroll
    for (int i = 0; i < 8; ++i) {
      float vv = acc[i] + bv;
      if (RELU) vv = fmaxf(vv, 0.f);
      size_t off = (size_t)(rowBase + moff + i) * N + ct * 16 + n;
      if (C16F) C16[off] = (_Float16)vv; else C32[off] = vv;
    }
  }
}

// ---------------- fused te-GRU step: GH = h @ WhhT (WMMA) + gates ----------------
// grid = 240 (d = blk/120, rowTile = blk%120), block 256.
__global__ void gru_step(const _Float16* Hf16, const float* Hf32,
                         const float* MW_f, const float* MW_b,
                         const _Float16* WhhP_f, const _Float16* WhhP_b,
                         const float* bhh_f, const float* bhh_b,
                         _Float16* Gflat, _Float16* Hf16o, float* Hf32o, int step) {
  __shared__ _Float16 As[16 * Hn];
  __shared__ float GH[16 * G3H];
  const int tid = threadIdx.x;
  const int d = blockIdx.x / 120;
  const int rowBase = (blockIdx.x % 120) * 16;
  const int l = (d == 0) ? step : (Ln - 1 - step);
  const _Float16* Hin  = Hf16 + (size_t)d * Bte * Hn;
  const float*    Hp32 = Hf32 + (size_t)d * Bte * Hn;
  const _Float16* Bw   = (d == 0) ? WhhP_f : WhhP_b;
  const float*    MW   = (d == 0) ? MW_f : MW_b;
  const float*    bhh  = (d == 0) ? bhh_f : bhh_b;

  // async-stage the 16x256 f16 H tile (8 KB contiguous) into LDS
  {
    unsigned ldsb = (unsigned)(uintptr_t)(&As[0]);
    const char* g = (const char*)(Hin + (size_t)rowBase * Hn);
    async_b128_to_lds(ldsb + tid * 16, g + tid * 16);
    async_b128_to_lds(ldsb + 4096 + tid * 16, g + 4096 + tid * 16);
    wait_async0();
  }
  __syncthreads();

  const int lane = tid & 31, wave = tid >> 5;
  const int KT = Hn >> 5;                         // 8
  for (int ct = wave; ct < (G3H >> 4); ct += 8) {
    v8f acc = {0.f, 0.f, 0.f, 0.f, 0.f, 0.f, 0.f, 0.f};
    for (int kt = 0; kt < KT; ++kt) {
      v16h a;
      load_a_frag(As, Hn, kt * 32, lane, a);
      v16h b = load_b_packed(Bw, KT, kt, ct, lane);
      acc = __builtin_amdgcn_wmma_f32_16x16x32_f16(false, a, false, b, (short)0, acc, false, false);
    }
    const int n = lane & 15;
    const int moff = (lane < 16) ? 0 : 8;
#pragma unroll
    for (int i = 0; i < 8; ++i) GH[(moff + i) * G3H + ct * 16 + n] = acc[i];
  }
  __syncthreads();

  for (int it = 0; it < 16; ++it) {
    const int lr = it, c = tid;                    // c in 0..255 (coalesced)
    const int j = rowBase + lr;
    const int i = j / Nn, n = j - i * Nn;
    const int lidx = (i + l) % Ln;
    const float* xw = MW + (size_t)(n * Ln + lidx) * G3H;
    const float ghr = GH[lr * G3H + c]          + bhh[c];
    const float ghz = GH[lr * G3H + Hn + c]     + bhh[Hn + c];
    const float ghn = GH[lr * G3H + 2 * Hn + c] + bhh[2 * Hn + c];
    const float r  = sigm(xw[c] + ghr);
    const float z  = sigm(xw[Hn + c] + ghz);
    const float nn = tanhf(xw[2 * Hn + c] + r * ghn);
    const float hp = Hp32[(size_t)j * Hn + c];
    const float h  = (1.f - z) * nn + z * hp;
    const size_t rr = (((size_t)l * Ln + i) * Nn + n) * 2 + d;
    Gflat[rr * Hn + c] = (_Float16)h;
    Hf16o[((size_t)d * Bte + j) * Hn + c] = (_Float16)h;
    Hf32o[((size_t)d * Bte + j) * Hn + c] = h;
  }
}

// ---------------- small helper kernels ----------------
__global__ void gather_m(const float* embed, const int* lines, float* Mflat) {
  int idx = blockIdx.x * 256 + threadIdx.x;
  if (idx >= Bte * Hn) return;
  int row = idx >> 8, c = idx & 255;
  Mflat[idx] = embed[(size_t)lines[row] * Hn + c];
}
__global__ void m2_out_kernel(const _Float16* U1, const float* W2, const float* b2, float* Gs) {
  size_t tid = (size_t)blockIdx.x * 256 + threadIdx.x;
  if (tid >= (size_t)GR * NOn) return;
  size_t rr = tid / NOn; int o = (int)(tid % NOn);
  float s = b2[o];
  const _Float16* row = U1 + rr * Hn;
  for (int c = 0; c < Hn; ++c) s += (float)row[c] * W2[c * NOn + o];
  Gs[tid] = sigm(s);
}
__global__ void p_kernel(const float* Gs, float* P) {
  int tid = blockIdx.x * 256 + threadIdx.x;
  if (tid >= Ln * Nn * 2 * NOn) return;
  int o = tid % NOn;
  int c = (tid / NOn) % 2;
  int n = (tid / (NOn * 2)) % Nn;
  int i = tid / (NOn * 2 * Nn);
  float* pr = P + (((size_t)i * Nn + n) * 2 * NOn + (c == 0 ? o : NOn + o)) * (2 * Ln);
  if (c == 0) { for (int j = 0; j < Ln; ++j) pr[j] = 0.f; }
  else        { for (int j = Ln; j < 2 * Ln; ++j) pr[j] = 0.f; }
  float cum = 1.f;
  for (int lp = 0; lp < Ln - 1; ++lp) {
    int l = lp + 1;
    float g = Gs[((((size_t)l * Ln + i) * Nn + n) * 2 + c) * NOn + o];
    float pfb = cum * g;
    int s = lp + 1;
    if (c == 0) pr[Ln + s] = pfb; else pr[Ln - 1 - s] = pfb;
    cum *= (1.f - g);
  }
  if (c == 0) pr[Ln] = cum; else pr[Ln - 1] = cum;
}
__global__ void init_h(const float* rnn_hxs, float* Hmain) {
  int idx = blockIdx.x * 256 + threadIdx.x;
  if (idx >= Nn * Hn) return;
  int n = idx >> 8, c = idx & 255;
  Hmain[idx] = rnn_hxs[(size_t)n * 340 + 4 + c];
}
__global__ void build_x(const float* cond, const float* Mflat, const int* active, float* X, int t) {
  int idx = blockIdx.x * 256 + threadIdx.x;
  if (idx >= Nn * (Cn + Hn)) return;
  int n = idx / (Cn + Hn), c = idx - n * (Cn + Hn);
  if (c < Cn) X[idx] = cond[((size_t)t * Nn + n) * Cn + c];
  else {
    int w = active[t * Nn + n];
    X[idx] = Mflat[((size_t)n * Ln + w) * Hn + (c - Cn)];
  }
}
__global__ void gate2(const float* XW, const float* GHm, float* Hmain, float* out, int t) {
  int idx = blockIdx.x * 256 + threadIdx.x;
  if (idx >= Nn * Hn) return;
  int n = idx >> 8, c = idx & 255;
  const float* xw = XW  + (size_t)n * G3H;
  const float* gh = GHm + (size_t)n * G3H;
  float r  = sigm(xw[c] + gh[c]);
  float z  = sigm(xw[Hn + c] + gh[Hn + c]);
  float nn = tanhf(xw[2 * Hn + c] + r * gh[2 * Hn + c]);
  float hp = Hmain[idx];
  float h  = (1.f - z) * nn + z * hp;
  Hmain[idx] = h;
  out[((size_t)t * Nn + n) * 340 + 4 + c] = h;
}
__global__ void head_kernel(const float* U2, const float* Hmain, const float* P,
                            const float* W3, const float* b3, const float* crW, const float* crb,
                            const float* rnn_hxs, const int* active, const int* actions,
                            float* out, int t) {
  __shared__ float u2s[Hn];
  __shared__ float logits[2 * NOn];
  __shared__ float osm[2 * NOn];
  __shared__ float vsh;
  const int n = blockIdx.x, tid = threadIdx.x; // 64 threads
  for (int c = tid; c < Hn; c += 64) u2s[c] = U2[(size_t)n * Hn + c];
  __syncthreads();
  if (tid < 2 * NOn) {
    float s = b3[tid];
    for (int c = 0; c < Hn; ++c) s += u2s[c] * W3[c * (2 * NOn) + tid];
    logits[tid] = s;
  }
  if (tid == 32) {
    float s = crb[0];
    for (int c = 0; c < Hn; ++c) s += Hmain[(size_t)n * Hn + c] * crW[c];
    vsh = s;
  }
  __syncthreads();
  if (tid == 0) {
    float mx = logits[0];
    for (int k = 1; k < 2 * NOn; ++k) mx = fmaxf(mx, logits[k]);
    float ss = 0.f;
    for (int k = 0; k < 2 * NOn; ++k) { float e = __expf(logits[k] - mx); osm[k] = e; ss += e; }
    for (int k = 0; k < 2 * NOn; ++k) osm[k] /= ss;
  }
  __syncthreads();
  const int w = active[t * Nn + n];
  float* o = out + ((size_t)t * Nn + n) * 340;
  if (tid < NAn) {
    float s = 0.f;
    for (int k = 0; k < 2 * NOn; ++k)
      s += osm[k] * P[(((size_t)w * Nn + n) * 2 * NOn + k) * (2 * Ln) + tid];
    o[260 + tid] = s;
    o[300 + tid] = rnn_hxs[(size_t)n * 340 + 300 + tid];
  }
  if (tid == 0) {
    o[0] = (float)actions[(t * Nn + n) * 2 + 0];
    o[1] = (float)actions[(t * Nn + n) * 2 + 1];
    o[2] = (float)w;
    o[3] = vsh;
  }
}

// ---------------- host ----------------
extern "C" void kernel_launch(void* const* d_in, const int* in_sizes, int n_in,
                              void* d_out, int out_size, void* d_ws, size_t ws_size,
                              hipStream_t stream) {
  const float* condition = (const float*)d_in[0];
  const float* rnn_hxs   = (const float*)d_in[1];
  const float* embed     = (const float*)d_in[2];
  const float* gc_Wih    = (const float*)d_in[3];
  const float* gc_Whh    = (const float*)d_in[4];
  const float* gc_bih    = (const float*)d_in[5];
  const float* gc_bhh    = (const float*)d_in[6];
  const float* teWih_f   = (const float*)d_in[7];
  const float* teWhh_f   = (const float*)d_in[8];
  const float* tebih_f   = (const float*)d_in[9];
  const float* tebhh_f   = (const float*)d_in[10];
  const float* teWih_b   = (const float*)d_in[11];
  const float* teWhh_b   = (const float*)d_in[12];
  const float* tebih_b   = (const float*)d_in[13];
  const float* tebhh_b   = (const float*)d_in[14];
  const float* mlp_W1    = (const float*)d_in[15];
  const float* mlp_b1    = (const float*)d_in[16];
  const float* mlp_W2    = (const float*)d_in[17];
  const float* mlp_b2    = (const float*)d_in[18];
  const float* mlp_W3    = (const float*)d_in[19];
  const float* mlp_b3    = (const float*)d_in[20];
  const float* m2_W1     = (const float*)d_in[21];
  const float* m2_b1     = (const float*)d_in[22];
  const float* m2_W2     = (const float*)d_in[23];
  const float* m2_b2     = (const float*)d_in[24];
  const float* cr_W      = (const float*)d_in[25];
  const float* cr_b      = (const float*)d_in[26];
  const int*   lines     = (const int*)d_in[27];
  const int*   active    = (const int*)d_in[28];
  const int*   actions   = (const int*)d_in[29];
  float* out = (float*)d_out;

  // bump allocator over workspace
  char* wp = (char*)d_ws;
  auto alloc = [&](size_t bytes) -> void* {
    void* r = (void*)wp;
    wp += (bytes + 255) & ~(size_t)255;
    return r;
  };
  _Float16* WihP_f  = (_Float16*)alloc((size_t)Hn * G3H * 2);           // packed B fragments
  _Float16* WhhP_f  = (_Float16*)alloc((size_t)Hn * G3H * 2);
  _Float16* WihP_b  = (_Float16*)alloc((size_t)Hn * G3H * 2);
  _Float16* WhhP_b  = (_Float16*)alloc((size_t)Hn * G3H * 2);
  _Float16* gcWihP  = (_Float16*)alloc((size_t)(Cn + Hn) * G3H * 2);
  _Float16* gcWhhP  = (_Float16*)alloc((size_t)Hn * G3H * 2);
  _Float16* m2W1P   = (_Float16*)alloc((size_t)Hn * Hn * 2);
  _Float16* mlpW1P  = (_Float16*)alloc((size_t)Hn * Hn * 2);
  _Float16* mlpW2P  = (_Float16*)alloc((size_t)Hn * Hn * 2);
  float*    Mflat   = (float*)alloc((size_t)Bte * Hn * 4);
  float*    MW_f    = (float*)alloc((size_t)Bte * G3H * 4);
  float*    MW_b    = (float*)alloc((size_t)Bte * G3H * 4);
  _Float16* Hf16    = (_Float16*)alloc((size_t)2 * Bte * Hn * 2);
  float*    Hf32    = (float*)alloc((size_t)2 * Bte * Hn * 4);
  _Float16* Gflat   = (_Float16*)alloc((size_t)GR * Hn * 2);
  _Float16* U1m2    = (_Float16*)alloc((size_t)GR * Hn * 2);
  float*    Gs      = (float*)alloc((size_t)GR * NOn * 4);
  float*    Pbuf    = (float*)alloc((size_t)Ln * Nn * 2 * NOn * 2 * Ln * 4);
  float*    Xbuf    = (float*)alloc((size_t)Nn * (Cn + Hn) * 4);
  float*    XWb     = (float*)alloc((size_t)Nn * G3H * 4);
  float*    GHb     = (float*)alloc((size_t)Nn * G3H * 4);
  float*    Hmain   = (float*)alloc((size_t)Nn * Hn * 4);
  float*    U1b     = (float*)alloc((size_t)Nn * Hn * 4);
  float*    U2b     = (float*)alloc((size_t)Nn * Hn * 4);

  auto CEIL = [](int a, int b) { return (a + b - 1) / b; };

  // 1) weight packing into WMMA-fragment layout (f16)
  pack_b<true ><<<CEIL(Hn * G3H, 256), 256, 0, stream>>>(teWih_f, WihP_f, Hn, G3H);
  pack_b<true ><<<CEIL(Hn * G3H, 256), 256, 0, stream>>>(teWhh_f, WhhP_f, Hn, G3H);
  pack_b<true ><<<CEIL(Hn * G3H, 256), 256, 0, stream>>>(teWih_b, WihP_b, Hn, G3H);
  pack_b<true ><<<CEIL(Hn * G3H, 256), 256, 0, stream>>>(teWhh_b, WhhP_b, Hn, G3H);
  pack_b<true ><<<CEIL((Cn + Hn) * G3H, 256), 256, 0, stream>>>(gc_Wih, gcWihP, Cn + Hn, G3H);
  pack_b<true ><<<CEIL(Hn * G3H, 256), 256, 0, stream>>>(gc_Whh, gcWhhP, Hn, G3H);
  pack_b<false><<<CEIL(Hn * Hn, 256), 256, 0, stream>>>(m2_W1, m2W1P, Hn, Hn);
  pack_b<false><<<CEIL(Hn * Hn, 256), 256, 0, stream>>>(mlp_W1, mlpW1P, Hn, Hn);
  pack_b<false><<<CEIL(Hn * Hn, 256), 256, 0, stream>>>(mlp_W2, mlpW2P, Hn, Hn);

  // 2) M = embed_task[lines]  (1920 x 256)
  gather_m<<<CEIL(Bte * Hn, 256), 256, 0, stream>>>(embed, lines, Mflat);

  // 3) MW_{f,b} = M @ WihT + bih   (1920 x 768, K=256)  [WMMA]
  wmma_gemm<false, false, false><<<Bte / 16, 256, 0, stream>>>(
      Mflat, nullptr, WihP_f, tebih_f, MW_f, nullptr, Hn, G3H);
  wmma_gemm<false, false, false><<<Bte / 16, 256, 0, stream>>>(
      Mflat, nullptr, WihP_b, tebih_b, MW_b, nullptr, Hn, G3H);

  // 4) zero GRU state
  hipMemsetAsync(Hf16, 0, (size_t)2 * Bte * Hn * 2, stream);
  hipMemsetAsync(Hf32, 0, (size_t)2 * Bte * Hn * 4, stream);

  // 5) bidirectional te-GRU, fused GEMM+gates per step  [WMMA + async LDS staging]
  for (int s = 0; s < Ln; ++s)
    gru_step<<<240, 256, 0, stream>>>(Hf16, Hf32, MW_f, MW_b, WhhP_f, WhhP_b,
                                      tebhh_f, tebhh_b, Gflat, Hf16, Hf32, s);

  // 6) U1 = relu(G @ m2_W1 + b1)  (76800 x 256, K=256)  [WMMA, f16 in/out, async A staging]
  wmma_gemm<true, true, true><<<GR / 16, 256, 0, stream>>>(
      nullptr, Gflat, m2W1P, m2_b1, nullptr, U1m2, Hn, Hn);

  // 7) Gs = sigmoid(U1 @ m2_W2 + b2)  (N=5, VALU)
  m2_out_kernel<<<CEIL(GR * NOn, 256), 256, 0, stream>>>(U1m2, m2_W2, m2_b2, Gs);

  // 8) survival chains -> P (20,96,10,40)
  p_kernel<<<CEIL(Ln * Nn * 2 * NOn, 256), 256, 0, stream>>>(Gs, Pbuf);

  // 9) main scan
  init_h<<<CEIL(Nn * Hn, 256), 256, 0, stream>>>(rnn_hxs, Hmain);
  for (int t = 0; t < Tn; ++t) {
    build_x<<<CEIL(Nn * (Cn + Hn), 256), 256, 0, stream>>>(condition, Mflat, active, Xbuf, t);
    wmma_gemm<false, false, false><<<Nn / 16, 256, 0, stream>>>(
        Xbuf, nullptr, gcWihP, gc_bih, XWb, nullptr, Cn + Hn, G3H);
    wmma_gemm<false, false, false><<<Nn / 16, 256, 0, stream>>>(
        Hmain, nullptr, gcWhhP, gc_bhh, GHb, nullptr, Hn, G3H);
    gate2<<<CEIL(Nn * Hn, 256), 256, 0, stream>>>(XWb, GHb, Hmain, out, t);
    wmma_gemm<false, false, true><<<Nn / 16, 256, 0, stream>>>(
        Hmain, nullptr, mlpW1P, mlp_b1, U1b, nullptr, Hn, Hn);
    wmma_gemm<false, false, true><<<Nn / 16, 256, 0, stream>>>(
        U1b, nullptr, mlpW2P, mlp_b2, U2b, nullptr, Hn, Hn);
    head_kernel<<<Nn, 64, 0, stream>>>(U2b, Hmain, Pbuf, mlp_W3, mlp_b3, cr_W, cr_b,
                                       rnn_hxs, active, actions, out, t);
  }
}
